// SlotMemory_13761075216775
// MI455X (gfx1250) — matrix operand order repaired and checked by reading, not compiled, not run on hardware
//
#include <hip/hip_runtime.h>
#include <math.h>

// Problem dims (fixed by reference): B=4, T=4096, D=2048, N_SLOTS=4096, TOP_K=4
#define M_TOK   16384
#define D_DIM   2048
#define NSLOT   4096

typedef __bf16 bf16;
typedef __attribute__((ext_vector_type(16))) __bf16 v16bf;
typedef __attribute__((ext_vector_type(8)))  __bf16 v8bf;
typedef __attribute__((ext_vector_type(4)))  __bf16 v4bf;
typedef __attribute__((ext_vector_type(8)))  float  v8f;
typedef __attribute__((ext_vector_type(4)))  float  v4f;
typedef __attribute__((__vector_size__(4 * sizeof(int)))) int v4i;

// ---------------------------------------------------------------------------
// CDNA5 async global->LDS staging (ASYNCcnt-tracked, bypasses VGPRs).
// Builtin signature (from compiler diagnostic): (v4i AS1*, v4i AS3*, Imm off, Imm cpol)
// ---------------------------------------------------------------------------
#if __has_builtin(__builtin_amdgcn_global_load_async_to_lds_b128) && \
    __has_builtin(__builtin_amdgcn_s_wait_asynccnt)
#define USE_ASYNC_LDS 1
#else
#define USE_ASYNC_LDS 0
#endif

#define AS1 __attribute__((address_space(1)))
#define AS3 __attribute__((address_space(3)))

static __device__ inline void async_cp16(const void* g, void* l) {
#if USE_ASYNC_LDS
  __builtin_amdgcn_global_load_async_to_lds_b128(
      (AS1 v4i*)(uintptr_t)g,
      (AS3 v4i*)(unsigned int)(uintptr_t)l, 0, 0);
#else
  (void)g; (void)l;
#endif
}

static __device__ inline v8f zero_v8f() {
  v8f z;
#pragma unroll
  for (int i = 0; i < 8; ++i) z[i] = 0.0f;
  return z;
}

static __device__ inline v16bf cat16(v8bf lo, v8bf hi) {
  return __builtin_shufflevector(lo, hi, 0,1,2,3,4,5,6,7,8,9,10,11,12,13,14,15);
}

// A fragment (16x32 bf16, ISA 7.12.2): lane L<16 -> row M=L, elems K = {0..7, 16..23};
// lanes 16..31 -> row M=L-16, elems K = {8..15, 24..31}. h = lane>>4.
static __device__ inline v16bf load_fragA(const bf16* s, int row, int h, int stride) {
  const bf16* p = s + row * stride + h * 8;
  v8bf lo = *(const v8bf*)(p);
  v8bf hi = *(const v8bf*)(p + 16);
  return cat16(lo, hi);
}

// B fragment (32x16 bf16): lane L<16 -> col N=L, K = {0..7, 16..23}; lanes 16..31 -> col L-16, K = {8..15, 24..31}.
static __device__ inline v16bf load_fragB(const bf16* s, int col, int h, int stride) {
  const bf16* p = s + col * stride + h * 8;
  v8bf lo = *(const v8bf*)(p);
  v8bf hi = *(const v8bf*)(p + 16);
  return cat16(lo, hi);
}

// Sorted (descending) top-4 insert; strict '>' keeps first occurrence on ties
// (matches jax.lax.top_k given ascending scan order).
static __device__ inline void top4_insert(float v, int id, float tv[4], int ti[4]) {
  if (v > tv[3]) {
    if (v > tv[1]) {
      if (v > tv[0]) {
        tv[3]=tv[2]; ti[3]=ti[2]; tv[2]=tv[1]; ti[2]=ti[1];
        tv[1]=tv[0]; ti[1]=ti[0]; tv[0]=v;     ti[0]=id;
      } else {
        tv[3]=tv[2]; ti[3]=ti[2]; tv[2]=tv[1]; ti[2]=ti[1];
        tv[1]=v;     ti[1]=id;
      }
    } else {
      if (v > tv[2]) { tv[3]=tv[2]; ti[3]=ti[2]; tv[2]=v; ti[2]=id; }
      else           { tv[3]=v;     ti[3]=id; }
    }
  }
}

// ---------------------------------------------------------------------------
// f32 -> bf16 bulk convert (vectorized x4)
// ---------------------------------------------------------------------------
__global__ __launch_bounds__(256) void f32_to_bf16_kernel(const float* __restrict__ src,
                                                          bf16* __restrict__ dst, int n4) {
  int i = blockIdx.x * 256 + threadIdx.x;
  if (i < n4) {
    v4f v = *(const v4f*)(src + (size_t)i * 4);
    v4bf o;
#pragma unroll
    for (int e = 0; e < 4; ++e) o[e] = (bf16)v[e];
    *(v4bf*)(dst + (size_t)i * 4) = o;
  }
}

// ---------------------------------------------------------------------------
// Tiled bf16 GEMM: C[M,N] = A[M,K] * B[N,K]^T  (row-major, contract last dim)
// 256 threads = 8 waves (2x4); WG tile 128x256; wave tile 64x64; K-step 32.
// Double-buffered LDS fed by async global->LDS copies when available.
// ---------------------------------------------------------------------------
#define LDT 40          // padded LDS stride in halves (80B -> conflict-free b128 reads)
#define ASZ (128 * LDT) // A buffer size (halves)
#define BSZ (256 * LDT) // B buffer size (halves)

template <typename OutT>
__global__ __launch_bounds__(256) void gemm_bf16_kernel(const bf16* __restrict__ A,
                                                        const bf16* __restrict__ B,
                                                        OutT* __restrict__ C,
                                                        int M, int N, int K) {
  __shared__ __attribute__((aligned(16))) bf16 sA[2 * ASZ];
  __shared__ __attribute__((aligned(16))) bf16 sB[2 * BSZ];

  const int tid   = threadIdx.x;
  const int lane  = tid & 31;
  const int w     = tid >> 5;
  const int wm    = w >> 2;           // 0..1
  const int wn    = w & 3;            // 0..3
  const int laneR = lane & 15;
  const int laneH = lane >> 4;
  const int bm = blockIdx.y, bn = blockIdx.x;

  // Per-thread staging coordinates (fixed for the whole kernel):
  //   A: one row, 16 contiguous halves;  B: 4 rows (stride 64), 8 halves each.
  const int rA = tid >> 1, cA = (tid & 1) * 16;
  const int rB = tid >> 2, cB = (tid & 3) * 8;
  const bf16* gA = A + (size_t)(bm * 128 + rA) * K + cA;
  const bf16* gB = B + (size_t)(bn * 256 + rB) * K + cB;
  bf16* lA = sA + rA * LDT + cA;
  bf16* lB = sB + rB * LDT + cB;

  v8f acc[4][4];
#pragma unroll
  for (int mi = 0; mi < 4; ++mi)
#pragma unroll
    for (int ni = 0; ni < 4; ++ni) acc[mi][ni] = zero_v8f();

  const int kTiles = K >> 5;

#if USE_ASYNC_LDS
  auto stage = [&](int kt, int buf) {
    const bf16* ga = gA + kt * 32;
    bf16* la = lA + buf * ASZ;
    async_cp16(ga, la);
    async_cp16(ga + 8, la + 8);
#pragma unroll
    for (int j = 0; j < 4; ++j)
      async_cp16(gB + (size_t)(64 * j) * K + kt * 32, lB + buf * BSZ + (64 * j) * LDT);
  };
  stage(0, 0);
  for (int kt = 0; kt < kTiles; ++kt) {
    if (kt + 1 < kTiles) {               // overlap next tile's copies with this compute
      stage(kt + 1, (kt + 1) & 1);
      __builtin_amdgcn_s_wait_asynccnt(6);   // 6 outstanding = tile kt+1 only
    } else {
      __builtin_amdgcn_s_wait_asynccnt(0);
    }
    __syncthreads();
    const bf16* pA = sA + (kt & 1) * ASZ;
    const bf16* pB = sB + (kt & 1) * BSZ;
#else
  for (int kt = 0; kt < kTiles; ++kt) {
    const int kb = kt << 5;
    *(v8bf*)(lA)     = *(const v8bf*)(gA + kb);
    *(v8bf*)(lA + 8) = *(const v8bf*)(gA + kb + 8);
#pragma unroll
    for (int j = 0; j < 4; ++j)
      *(v8bf*)(lB + (64 * j) * LDT) = *(const v8bf*)(gB + (size_t)(64 * j) * K + kb);
    __syncthreads();
    if (kt + 1 < kTiles) {
      __builtin_prefetch(gA + kb + 32, 0, 0);
      __builtin_prefetch(gB + kb + 32, 0, 0);
    }
    const bf16* pA = sA;
    const bf16* pB = sB;
#endif
    v16bf a[4], b[4];
#pragma unroll
    for (int mi = 0; mi < 4; ++mi)
      a[mi] = load_fragA(pA, wm * 64 + mi * 16 + laneR, laneH, LDT);
#pragma unroll
    for (int ni = 0; ni < 4; ++ni)
      b[ni] = load_fragB(pB, wn * 64 + ni * 16 + laneR, laneH, LDT);

#pragma unroll
    for (int mi = 0; mi < 4; ++mi)
#pragma unroll
      for (int ni = 0; ni < 4; ++ni)
        acc[mi][ni] = __builtin_amdgcn_wmma_f32_16x16x32_bf16(
            false, a[mi], false, b[ni], (short)0, acc[mi][ni], false, false);
    __syncthreads();
  }

  // Epilogue: C/D layout — VGPR r holds M = r + 8*(lane>>4), N = lane&15.
#pragma unroll
  for (int mi = 0; mi < 4; ++mi)
#pragma unroll
    for (int ni = 0; ni < 4; ++ni) {
      int col = bn * 256 + wn * 64 + ni * 16 + laneR;
#pragma unroll
      for (int r = 0; r < 8; ++r) {
        int row = bm * 128 + wm * 64 + mi * 16 + r + 8 * laneH;
        C[(size_t)row * N + col] = (OutT)acc[mi][ni][r];
      }
    }
}

// ---------------------------------------------------------------------------
// Fused scores + top-k + softmax:
//   scores[m, n] = scale * dot(Q[m,:], K[n,:]);  running top-4 per row.
// 128 threads = 4 waves (1x4); M-tile 64, N in chunks of 256; scores stay in LDS.
// ---------------------------------------------------------------------------
#define SSTR 257            // f32 score stride (257 % 64 == 1 -> conflict-free scan)
#define TASZ (64 * LDT)
#define TBSZ (256 * LDT)

__global__ __launch_bounds__(128) void scores_topk_kernel(const bf16* __restrict__ Q,
                                                          const bf16* __restrict__ Kb,
                                                          float* __restrict__ attn,
                                                          int* __restrict__ ids,
                                                          int K, float scale) {
  __shared__ __attribute__((aligned(16))) bf16 sA[2 * TASZ];
  __shared__ __attribute__((aligned(16))) bf16 sB[2 * TBSZ];
  __shared__ float sS[64 * SSTR];
  __shared__ float mtv[64][4];
  __shared__ int   mti[64][4];

  const int tid   = threadIdx.x;
  const int lane  = tid & 31;
  const int w     = tid >> 5;         // 0..3 -> wave N offset
  const int laneR = lane & 15;
  const int laneH = lane >> 4;
  const int bm    = blockIdx.x;       // 64-row tile index

  // Staging coords: A: row rA, 16 contiguous halves; B: 8 rows (stride 32), 8 halves.
  const int rA = tid >> 1, cA = (tid & 1) * 16;
  const int rB = tid >> 2, cB = (tid & 3) * 8;
  const bf16* gA = Q + (size_t)(bm * 64 + rA) * K + cA;
  bf16* lA = sA + rA * LDT + cA;
  bf16* lB = sB + rB * LDT + cB;

  const int rowL = tid & 63;          // row this thread scans
  const int seg  = tid >> 6;          // 0/1: which 128-col half of the chunk

  float tv[4] = {-3.4e38f, -3.4e38f, -3.4e38f, -3.4e38f};
  int   ti[4] = {0, 0, 0, 0};

  const int kTiles = K >> 5;
  for (int nc = 0; nc < NSLOT / 256; ++nc) {
    const bf16* gB = Kb + (size_t)(nc * 256 + rB) * K + cB;

    v8f acc[4][4];
#pragma unroll
    for (int mi = 0; mi < 4; ++mi)
#pragma unroll
      for (int ni = 0; ni < 4; ++ni) acc[mi][ni] = zero_v8f();

#if USE_ASYNC_LDS
    auto stage = [&](int kt, int buf) {
      const bf16* ga = gA + kt * 32;
      bf16* la = lA + buf * TASZ;
      async_cp16(ga, la);
      async_cp16(ga + 8, la + 8);
#pragma unroll
      for (int j = 0; j < 8; ++j)
        async_cp16(gB + (size_t)(32 * j) * K + kt * 32, lB + buf * TBSZ + (32 * j) * LDT);
    };
    stage(0, 0);
    for (int kt = 0; kt < kTiles; ++kt) {
      if (kt + 1 < kTiles) {
        stage(kt + 1, (kt + 1) & 1);
        __builtin_amdgcn_s_wait_asynccnt(10);  // 10 outstanding = tile kt+1 only
      } else {
        __builtin_amdgcn_s_wait_asynccnt(0);
      }
      __syncthreads();
      const bf16* pA = sA + (kt & 1) * TASZ;
      const bf16* pB = sB + (kt & 1) * TBSZ;
#else
    for (int kt = 0; kt < kTiles; ++kt) {
      const int kb = kt << 5;
      *(v8bf*)(lA)     = *(const v8bf*)(gA + kb);
      *(v8bf*)(lA + 8) = *(const v8bf*)(gA + kb + 8);
#pragma unroll
      for (int j = 0; j < 8; ++j)
        *(v8bf*)(lB + (32 * j) * LDT) = *(const v8bf*)(gB + (size_t)(32 * j) * K + kb);
      __syncthreads();
      if (kt + 1 < kTiles) {
        __builtin_prefetch(gA + kb + 32, 0, 0);
        __builtin_prefetch(gB + kb + 32, 0, 0);
      }
      const bf16* pA = sA;
      const bf16* pB = sB;
#endif
      v16bf a[4], b[4];
#pragma unroll
      for (int mi = 0; mi < 4; ++mi)
        a[mi] = load_fragA(pA, mi * 16 + laneR, laneH, LDT);
#pragma unroll
      for (int ni = 0; ni < 4; ++ni)
        b[ni] = load_fragB(pB, w * 64 + ni * 16 + laneR, laneH, LDT);

#pragma unroll
      for (int mi = 0; mi < 4; ++mi)
#pragma unroll
        for (int ni = 0; ni < 4; ++ni)
          acc[mi][ni] = __builtin_amdgcn_wmma_f32_16x16x32_bf16(
              false, a[mi], false, b[ni], (short)0, acc[mi][ni], false, false);
      __syncthreads();
    }

    // Spill scaled scores to LDS (64 x 256 f32).
#pragma unroll
    for (int mi = 0; mi < 4; ++mi)
#pragma unroll
      for (int ni = 0; ni < 4; ++ni) {
        int colL = w * 64 + ni * 16 + laneR;
#pragma unroll
        for (int r = 0; r < 8; ++r) {
          int rL = mi * 16 + r + 8 * laneH;
          sS[rL * SSTR + colL] = scale * acc[mi][ni][r];
        }
      }
    __syncthreads();

    // Running top-4 scan: thread scans 128 cols of its row.
    const int cbase = seg * 128;
    for (int i = 0; i < 128; ++i) {
      float v = sS[rowL * SSTR + cbase + i];
      top4_insert(v, nc * 256 + cbase + i, tv, ti);
    }
    __syncthreads();
  }

  // Merge the two per-row partial top-4 sets, then softmax.
  if (seg == 1) {
#pragma unroll
    for (int j = 0; j < 4; ++j) { mtv[rowL][j] = tv[j]; mti[rowL][j] = ti[j]; }
  }
  __syncthreads();
  if (seg == 0) {
#pragma unroll
    for (int j = 0; j < 4; ++j) top4_insert(mtv[rowL][j], mti[rowL][j], tv, ti);
    float e[4], s = 0.0f;
#pragma unroll
    for (int j = 0; j < 4; ++j) { e[j] = __expf(tv[j] - tv[0]); s += e[j]; }
    float inv = 1.0f / s;
    int grow = bm * 64 + rowL;
#pragma unroll
    for (int j = 0; j < 4; ++j) {
      attn[grow * 4 + j] = e[j] * inv;
      ids[grow * 4 + j]  = ti[j];
    }
  }
}

// ---------------------------------------------------------------------------
// retrieved[m,:] = sum_j attn[m,j] * slot_values[ids[m,j], :]  -> bf16
// ---------------------------------------------------------------------------
__global__ __launch_bounds__(256) void retrieve_kernel(const float* __restrict__ V,
                                                       const float* __restrict__ attn,
                                                       const int* __restrict__ ids,
                                                       bf16* __restrict__ R) {
  const int m = blockIdx.x;
  float a0 = attn[m * 4 + 0], a1 = attn[m * 4 + 1], a2 = attn[m * 4 + 2], a3 = attn[m * 4 + 3];
  const float* v0 = V + (size_t)ids[m * 4 + 0] * D_DIM;
  const float* v1 = V + (size_t)ids[m * 4 + 1] * D_DIM;
  const float* v2 = V + (size_t)ids[m * 4 + 2] * D_DIM;
  const float* v3 = V + (size_t)ids[m * 4 + 3] * D_DIM;

  for (int d = threadIdx.x * 4; d < D_DIM; d += 256 * 4) {
    v4f s = a0 * (*(const v4f*)(v0 + d)) + a1 * (*(const v4f*)(v1 + d)) +
            a2 * (*(const v4f*)(v2 + d)) + a3 * (*(const v4f*)(v3 + d));
    v4bf o;
#pragma unroll
    for (int e = 0; e < 4; ++e) o[e] = (bf16)s[e];
    *(v4bf*)(R + (size_t)m * D_DIM + d) = o;
  }
}

// ---------------------------------------------------------------------------
// Host launcher
// ---------------------------------------------------------------------------
extern "C" void kernel_launch(void* const* d_in, const int* in_sizes, int n_in,
                              void* d_out, int out_size, void* d_ws, size_t ws_size,
                              hipStream_t stream) {
  (void)in_sizes; (void)n_in; (void)out_size; (void)ws_size;

  const float* x   = (const float*)d_in[0];  // [16384, 2048]
  const float* sk  = (const float*)d_in[1];  // [4096, 2048]
  const float* sv  = (const float*)d_in[2];  // [4096, 2048]
  const float* Wq  = (const float*)d_in[3];  // [2048, 2048]
  const float* Wo  = (const float*)d_in[4];  // [2048, 2048]
  // d_in[5] = top_k (== 4, hardcoded)

  char* ws = (char*)d_ws;
  size_t off = 0;
  bf16* xb    = (bf16*)(ws + off); off += (size_t)M_TOK * D_DIM * 2;   // 64 MB
  bf16* qb    = (bf16*)(ws + off); off += (size_t)M_TOK * D_DIM * 2;   // 64 MB
  bf16* keysb = (bf16*)(ws + off); off += (size_t)NSLOT * D_DIM * 2;   // 16 MB
  bf16* wqb   = (bf16*)(ws + off); off += (size_t)D_DIM * D_DIM * 2;   //  8 MB
  bf16* wob   = (bf16*)(ws + off); off += (size_t)D_DIM * D_DIM * 2;   //  8 MB
  bf16* retb  = (bf16*)(ws + off); off += (size_t)M_TOK * D_DIM * 2;   // 64 MB
  float* attn = (float*)(ws + off); off += (size_t)M_TOK * 4 * 4;
  int*   tids = (int*)(ws + off);   off += (size_t)M_TOK * 4 * 4;

  // 1) bf16 conversions (one streaming pass each)
  f32_to_bf16_kernel<<<(M_TOK * D_DIM / 4) / 256, 256, 0, stream>>>(x,  xb,    M_TOK * D_DIM / 4);
  f32_to_bf16_kernel<<<(NSLOT * D_DIM / 4) / 256, 256, 0, stream>>>(sk, keysb, NSLOT * D_DIM / 4);
  f32_to_bf16_kernel<<<(D_DIM * D_DIM / 4) / 256, 256, 0, stream>>>(Wq, wqb,   D_DIM * D_DIM / 4);
  f32_to_bf16_kernel<<<(D_DIM * D_DIM / 4) / 256, 256, 0, stream>>>(Wo, wob,   D_DIM * D_DIM / 4);

  // 2) q = x * Wq^T   (bf16 out, feeds score GEMM)
  gemm_bf16_kernel<bf16><<<dim3(D_DIM / 256, M_TOK / 128), 256, 0, stream>>>(
      xb, wqb, qb, M_TOK, D_DIM, D_DIM);

  // 3) fused scores + top-4 + softmax (scores stay in LDS)
  const float scale = 0.022097086912079608f;  // 1/sqrt(2048)
  scores_topk_kernel<<<M_TOK / 64, 128, 0, stream>>>(qb, keysb, attn, tids, D_DIM, scale);

  // 4) retrieved = sum_k attn_k * V[id_k]
  retrieve_kernel<<<M_TOK, 256, 0, stream>>>(sv, attn, tids, retb);

  // 5) out = retrieved * Wout^T (f32 out)
  gemm_bf16_kernel<float><<<dim3(D_DIM / 256, M_TOK / 128), 256, 0, stream>>>(
      retb, wob, (float*)d_out, M_TOK, D_DIM, D_DIM);
}
